// EdgeConvGCNSegmentation_61641370632421
// MI455X (gfx1250) — compile-verified, matrix operand order checked by reading, not run on hardware
//
#include <hip/hip_runtime.h>
#include <cstdint>
#include <cstddef>

#define KNN 20
#define TB 256
#define BNEPS 1e-5f

typedef __attribute__((ext_vector_type(16))) _Float16 v16h;
typedef __attribute__((ext_vector_type(8)))  _Float16 v8h;
typedef __attribute__((ext_vector_type(8)))  float    v8f;

// ---------------------------------------------------------------- utilities
__global__ void k_zero(float* __restrict__ p, int n) {
  int i = blockIdx.x * blockDim.x + threadIdx.x;
  if (i < n) p[i] = 0.f;
}

// ---------------------------------------------------------------- GraphConv
__global__ void k_deg(const int* __restrict__ src, const int* __restrict__ dst,
                      float* __restrict__ degO, float* __restrict__ degI, int e) {
  int i = blockIdx.x * blockDim.x + threadIdx.x;
  if (i < e) {
    atomicAdd(&degO[src[i]], 1.f);
    atomicAdd(&degI[dst[i]], 1.f);
  }
}

__global__ void k_deg_inv(float* __restrict__ degO, float* __restrict__ degI, int n) {
  int i = blockIdx.x * blockDim.x + threadIdx.x;
  if (i < n) {
    degO[i] = rsqrtf(fmaxf(degO[i], 1.f));
    degI[i] = rsqrtf(fmaxf(degI[i], 1.f));
  }
}

// agg[dst, c] += x[src, c] * degO^-1/2[src]   (one thread per edge*channel)
__global__ void k_gcn_scatter(const float* __restrict__ x, const int* __restrict__ src,
                              const int* __restrict__ dst, const float* __restrict__ degO,
                              float* __restrict__ agg, int e, int in) {
  int i = blockIdx.x * blockDim.x + threadIdx.x;
  if (i < e * in) {
    int ed = i / in, c = i - ed * in;
    int s = src[ed], d = dst[ed];
    atomicAdd(&agg[(size_t)d * in + c], x[(size_t)s * in + c] * degO[s]);
  }
}

// out[i,o] = relu( (agg[i,:] * degI^-1/2[i]) @ W + b )
__global__ void k_gcn_linear(const float* __restrict__ agg, const float* __restrict__ degI,
                             const float* __restrict__ W, const float* __restrict__ b,
                             float* __restrict__ out, int n, int in, int outd, int relu) {
  int i = blockIdx.x * blockDim.x + threadIdx.x;
  if (i >= n * outd) return;
  int node = i / outd, o = i - node * outd;
  float di = degI[node];
  float acc = b[o];
  for (int c = 0; c < in; ++c) acc += agg[(size_t)node * in + c] * di * W[c * outd + o];
  out[i] = relu ? fmaxf(acc, 0.f) : acc;
}

// ---------------------------------------------------------------- kNN (WMMA)
// Pack node features into [N][32] f16 (zero padded) + per-node squared norm.
// Note: the reference's global min-shift / max-scale is a uniform affine map,
// which rescales ALL pairwise distances by the same factor -> identical top-k.
__global__ void k_knn_prep(const float* __restrict__ x, int dim,
                           _Float16* __restrict__ xh, float* __restrict__ sq, int n) {
  int i = blockIdx.x * blockDim.x + threadIdx.x;
  if (i >= n) return;
  float s = 0.f;
  for (int c = 0; c < 32; ++c) {
    float v = (c < dim) ? x[(size_t)i * dim + c] : 0.f;
    _Float16 h = (_Float16)v;
    xh[(size_t)i * 32 + c] = h;
    float hv = (float)h;
    s += hv * hv;
  }
  sq[i] = s;
}

// Build a 16x32 f16 WMMA fragment row: lanes 0-15 hold K {0..7,16..23},
// lanes 16-31 hold K {8..15,24..31} (CDNA5 ISA 7.12.2 layout). koff = (lane>>4)*8.
__device__ inline v16h load_frag(const _Float16* __restrict__ xh, int node, int koff) {
  const _Float16* p = xh + (size_t)node * 32 + koff;
  v8h lo = *(const v8h*)(p);
  v8h hi = *(const v8h*)(p + 16);
  return __builtin_shufflevector(lo, hi, 0, 1, 2, 3, 4, 5, 6, 7,
                                 8, 9, 10, 11, 12, 13, 14, 15);
}

// Sorted (ascending) top-K insertion, BRANCHLESS ripple: each stage is
// min/max + cndmask pair with static register indices -> straight-line code,
// arrays stay in fixed registers (no exec-masked array copies).
__device__ inline void topk_insert(float (&bd)[KNN], int (&bi)[KNN], float d, int idx) {
  if (d < bd[KNN - 1]) {
    float cd = d;
    int   ci = idx;
#pragma unroll
    for (int k = 0; k < KNN; ++k) {
      bool  lt = cd < bd[k];
      float nd = lt ? cd : bd[k];
      int   ni = lt ? ci : bi[k];
      float xd = lt ? bd[k] : cd;
      int   xi = lt ? bi[k] : ci;
      bd[k] = nd;
      bi[k] = ni;
      cd = xd;
      ci = xi;
    }
  }
}

// One wave per 16-row tile; streams all column tiles, fused Gram-matrix (WMMA)
// -> d^2 -> in-register top-20 selection. Distance matrix never touches HBM.
__global__ __launch_bounds__(32) void k_knn(const _Float16* __restrict__ xh,
                                            const float* __restrict__ sq,
                                            int* __restrict__ col_out, int n) {
  __shared__ float tile[16 * 16];
  __shared__ float sqrow[16];
  __shared__ float mbd[16 * KNN];
  __shared__ int   mbi[16 * KNN];

  const int lane    = threadIdx.x;
  const int rowbase = blockIdx.x * 16;
  const int rhalf   = lane >> 4;    // 0 or 1
  const int rcol    = lane & 15;
  const int koff    = rhalf * 8;

  if (lane < 16) sqrow[lane] = sq[rowbase + lane];
  v16h a = load_frag(xh, rowbase + rcol, koff);   // hoisted A fragment

  float bd[KNN];
  int   bi[KNN];
#pragma unroll
  for (int k = 0; k < KNN; ++k) { bd[k] = 3.4e38f; bi[k] = 0; }

  asm volatile("s_wait_dscnt 0" ::: "memory");

  for (int cb = 0; cb < n; cb += 16) {
    if (cb + 16 < n) __builtin_prefetch(xh + (size_t)(cb + 16) * 32 + koff, 0, 1);
    v16h  b   = load_frag(xh, cb + rcol, koff);
    float sqc = sq[cb + rcol];

    v8f c = {};
    c = __builtin_amdgcn_wmma_f32_16x16x32_f16(false, a, false, b,
                                               (short)0, c, false, false);
#pragma unroll
    for (int v = 0; v < 8; ++v) {
      int m = v + 8 * rhalf;                       // C layout: row = v + 8*(lane>=16)
      tile[m * 16 + rcol] = sqrow[m] + sqc - 2.0f * c[v];
    }
    asm volatile("s_wait_dscnt 0" ::: "memory");   // wave-local store->load ordering

    // lane scans row (lane&15), columns koff..koff+7 of this tile
#pragma unroll
    for (int t = 0; t < 8; ++t) {
      float d = tile[rcol * 16 + koff + t];
      topk_insert(bd, bi, d, cb + koff + t);
    }
    asm volatile("s_wait_dscnt 0" ::: "memory");
  }

  // merge the two half-column scans per row (lane L and lane L+16)
  if (rhalf == 1) {
#pragma unroll
    for (int k = 0; k < KNN; ++k) {
      mbd[rcol * KNN + k] = bd[k];
      mbi[rcol * KNN + k] = bi[k];
    }
  }
  asm volatile("s_wait_dscnt 0" ::: "memory");
  __builtin_amdgcn_wave_barrier();
  if (rhalf == 0) {
#pragma unroll
    for (int k = 0; k < KNN; ++k)
      topk_insert(bd, bi, mbd[rcol * KNN + k], mbi[rcol * KNN + k]);
    int* out = col_out + (size_t)(rowbase + rcol) * KNN;
#pragma unroll
    for (int k = 0; k < KNN; ++k) out[k] = bi[k];
  }
}

// ---------------------------------------------------------------- EdgeConv
// layer1: feat=[x[col], x[row]-x[col]] @ W + b ; accumulate BN stats
template <int DIN, int D>
__global__ __launch_bounds__(TB) void k_edge_l1(const float* __restrict__ x,
                                                const int* __restrict__ colI,
                                                const float* __restrict__ W,
                                                const float* __restrict__ bia,
                                                float* __restrict__ y,
                                                float* __restrict__ s1,
                                                float* __restrict__ s2, int e2) {
  __shared__ float ssum[32], ssq[32];
  int tid = threadIdx.x;
  if (tid < 32) { ssum[tid] = 0.f; ssq[tid] = 0.f; }
  __syncthreads();
  int e = blockIdx.x * blockDim.x + tid;
  if (e < e2) {
    int r = e / KNN;
    int c = colI[e];
    float feat[D];
#pragma unroll
    for (int j = 0; j < DIN; ++j) {
      float xi = x[(size_t)c * DIN + j];
      float xj = x[(size_t)r * DIN + j];
      feat[j]       = xi;
      feat[DIN + j] = xj - xi;
    }
#pragma unroll
    for (int o = 0; o < D; ++o) {
      float acc = bia[o];
#pragma unroll
      for (int j = 0; j < D; ++j) acc += feat[j] * W[j * D + o];
      y[(size_t)e * D + o] = acc;
      atomicAdd(&ssum[o], acc);
      atomicAdd(&ssq[o], acc * acc);
    }
  }
  __syncthreads();
  if (tid < D) { atomicAdd(&s1[tid], ssum[tid]); atomicAdd(&s2[tid], ssq[tid]); }
}

// BN fold: a = g*rsqrt(var+eps), b = beta - mean*a   (training-mode, biased var)
__global__ void k_bn_fin(const float* __restrict__ s1, const float* __restrict__ s2,
                         const float* __restrict__ g, const float* __restrict__ be,
                         float* __restrict__ a, float* __restrict__ bsh,
                         int D, float inv_n) {
  int o = threadIdx.x;
  if (o >= D) return;
  float m = s1[o] * inv_n;
  float v = fmaxf(s2[o] * inv_n - m * m, 0.f);
  float sc = g[o] * rsqrtf(v + BNEPS);
  a[o] = sc;
  bsh[o] = be[o] - m * sc;
}

// layer2. APPLY=0: z = relu(BN1(y)) @ W + b, accumulate BN2 stats (z not stored).
// APPLY=1: recompute z, apply BN2+ReLU, segment-max into out[col] via uint atomicMax
// (valid: post-ReLU values are >= 0, and every node has a self edge).
template <int D, int APPLY>
__global__ __launch_bounds__(TB) void k_edge_l2(const float* __restrict__ y,
                                                const float* __restrict__ a1,
                                                const float* __restrict__ b1n,
                                                const float* __restrict__ W,
                                                const float* __restrict__ bia,
                                                float* __restrict__ s1,
                                                float* __restrict__ s2,
                                                const float* __restrict__ a2,
                                                const float* __restrict__ b2n,
                                                const int* __restrict__ colI,
                                                float* __restrict__ out, int e2) {
  __shared__ float ssum[32], ssq[32];
  int tid = threadIdx.x;
  if (!APPLY && tid < 32) { ssum[tid] = 0.f; ssq[tid] = 0.f; }
  __syncthreads();
  int e = blockIdx.x * blockDim.x + tid;
  if (e < e2) {
    float t[D];
#pragma unroll
    for (int j = 0; j < D; ++j)
      t[j] = fmaxf(y[(size_t)e * D + j] * a1[j] + b1n[j], 0.f);
    int c = APPLY ? colI[e] : 0;
#pragma unroll
    for (int o = 0; o < D; ++o) {
      float acc = bia[o];
#pragma unroll
      for (int j = 0; j < D; ++j) acc += t[j] * W[j * D + o];
      if (APPLY) {
        float val = fmaxf(acc * a2[o] + b2n[o], 0.f);
        atomicMax((unsigned int*)&out[(size_t)c * D + o], __float_as_uint(val));
      } else {
        atomicAdd(&ssum[o], acc);
        atomicAdd(&ssq[o], acc * acc);
      }
    }
  }
  __syncthreads();
  if (!APPLY && tid < D) { atomicAdd(&s1[tid], ssum[tid]); atomicAdd(&s2[tid], ssq[tid]); }
}

// ---------------------------------------------------------------- driver
extern "C" void kernel_launch(void* const* d_in, const int* in_sizes, int n_in,
                              void* d_out, int out_size, void* d_ws, size_t ws_size,
                              hipStream_t stream) {
  const float* features = (const float*)d_in[0];
  const int*   esrc = (const int*)d_in[1];
  const int*   edst = (const int*)d_in[2];
  const float* W1 = (const float*)d_in[3];  const float* b1 = (const float*)d_in[4];
  const float* W2 = (const float*)d_in[5];  const float* b2 = (const float*)d_in[6];
  const float* W3 = (const float*)d_in[7];  const float* b3 = (const float*)d_in[8];
  const float* W4 = (const float*)d_in[9];  const float* b4 = (const float*)d_in[10];
  const float* e1w0 = (const float*)d_in[11]; const float* e1b0 = (const float*)d_in[12];
  const float* e1g0 = (const float*)d_in[13]; const float* e1bb0 = (const float*)d_in[14];
  const float* e1w1 = (const float*)d_in[15]; const float* e1b1 = (const float*)d_in[16];
  const float* e1g1 = (const float*)d_in[17]; const float* e1bb1 = (const float*)d_in[18];
  const float* e2w0 = (const float*)d_in[19]; const float* e2b0 = (const float*)d_in[20];
  const float* e2g0 = (const float*)d_in[21]; const float* e2bb0 = (const float*)d_in[22];
  const float* e2w1 = (const float*)d_in[23]; const float* e2b1 = (const float*)d_in[24];
  const float* e2g1 = (const float*)d_in[25]; const float* e2bb1 = (const float*)d_in[26];
  (void)n_in; (void)out_size; (void)ws_size;

  const int N  = in_sizes[0] / 6;      // 16384
  const int E  = in_sizes[1];          // 98304
  const int E2 = N * KNN;              // 327680
  const float invE2 = 1.f / (float)E2;

  char* p = (char*)d_ws;
  auto take = [&](size_t bytes) -> char* {
    char* r = p;
    p += (bytes + 255) & ~(size_t)255;
    return r;
  };
  float*    degO = (float*)take((size_t)N * 4);
  float*    degI = (float*)take((size_t)N * 4);
  float*    agg  = (float*)take((size_t)N * 32 * 4);
  float*    hA   = (float*)take((size_t)N * 32 * 4);
  float*    hB   = (float*)take((size_t)N * 32 * 4);
  _Float16* xh   = (_Float16*)take((size_t)N * 32 * 2);
  float*    sq   = (float*)take((size_t)N * 4);
  int*      colI = (int*)take((size_t)E2 * 4);
  float*    s1   = (float*)take(32 * 4);
  float*    s2   = (float*)take(32 * 4);
  float*    a1   = (float*)take(32 * 4);
  float*    bn1  = (float*)take(32 * 4);
  float*    a2   = (float*)take(32 * 4);
  float*    bn2  = (float*)take(32 * 4);
  float*    y    = (float*)take((size_t)E2 * 32 * 4);

  auto nb = [](long long n) { return (int)((n + TB - 1) / TB); };

  // degrees (shared by all GraphConvs)
  k_zero<<<nb(N), TB, 0, stream>>>(degO, N);
  k_zero<<<nb(N), TB, 0, stream>>>(degI, N);
  k_deg<<<nb(E), TB, 0, stream>>>(esrc, edst, degO, degI, E);
  k_deg_inv<<<nb(N), TB, 0, stream>>>(degO, degI, N);

  // GCN1: 6 -> 8, relu
  k_zero<<<nb((long long)N * 6), TB, 0, stream>>>(agg, N * 6);
  k_gcn_scatter<<<nb((long long)E * 6), TB, 0, stream>>>(features, esrc, edst, degO, agg, E, 6);
  k_gcn_linear<<<nb((long long)N * 8), TB, 0, stream>>>(agg, degI, W1, b1, hA, N, 6, 8, 1);

  // kNN #1 on hA (dim 8)
  k_knn_prep<<<nb(N), TB, 0, stream>>>(hA, 8, xh, sq, N);
  k_knn<<<N / 16, 32, 0, stream>>>(xh, sq, colI, N);

  // EdgeConv1: din=8, D=16 -> hB
  k_zero<<<1, 64, 0, stream>>>(s1, 32);
  k_zero<<<1, 64, 0, stream>>>(s2, 32);
  k_edge_l1<8, 16><<<nb(E2), TB, 0, stream>>>(hA, colI, e1w0, e1b0, y, s1, s2, E2);
  k_bn_fin<<<1, 32, 0, stream>>>(s1, s2, e1g0, e1bb0, a1, bn1, 16, invE2);
  k_zero<<<1, 64, 0, stream>>>(s1, 32);
  k_zero<<<1, 64, 0, stream>>>(s2, 32);
  k_edge_l2<16, 0><<<nb(E2), TB, 0, stream>>>(y, a1, bn1, e1w1, e1b1, s1, s2,
                                              nullptr, nullptr, colI, nullptr, E2);
  k_bn_fin<<<1, 32, 0, stream>>>(s1, s2, e1g1, e1bb1, a2, bn2, 16, invE2);
  k_zero<<<nb((long long)N * 16), TB, 0, stream>>>(hB, N * 16);
  k_edge_l2<16, 1><<<nb(E2), TB, 0, stream>>>(y, a1, bn1, e1w1, e1b1, nullptr, nullptr,
                                              a2, bn2, colI, hB, E2);

  // GCN2: 16 -> 16, relu  (hB -> hA)
  k_zero<<<nb((long long)N * 16), TB, 0, stream>>>(agg, N * 16);
  k_gcn_scatter<<<nb((long long)E * 16), TB, 0, stream>>>(hB, esrc, edst, degO, agg, E, 16);
  k_gcn_linear<<<nb((long long)N * 16), TB, 0, stream>>>(agg, degI, W2, b2, hA, N, 16, 16, 1);

  // kNN #2 on hA (dim 16)
  k_knn_prep<<<nb(N), TB, 0, stream>>>(hA, 16, xh, sq, N);
  k_knn<<<N / 16, 32, 0, stream>>>(xh, sq, colI, N);

  // EdgeConv2: din=16, D=32 -> hB
  k_zero<<<1, 64, 0, stream>>>(s1, 32);
  k_zero<<<1, 64, 0, stream>>>(s2, 32);
  k_edge_l1<16, 32><<<nb(E2), TB, 0, stream>>>(hA, colI, e2w0, e2b0, y, s1, s2, E2);
  k_bn_fin<<<1, 32, 0, stream>>>(s1, s2, e2g0, e2bb0, a1, bn1, 32, invE2);
  k_zero<<<1, 64, 0, stream>>>(s1, 32);
  k_zero<<<1, 64, 0, stream>>>(s2, 32);
  k_edge_l2<32, 0><<<nb(E2), TB, 0, stream>>>(y, a1, bn1, e2w1, e2b1, s1, s2,
                                              nullptr, nullptr, colI, nullptr, E2);
  k_bn_fin<<<1, 32, 0, stream>>>(s1, s2, e2g1, e2bb1, a2, bn2, 32, invE2);
  k_zero<<<nb((long long)N * 32), TB, 0, stream>>>(hB, N * 32);
  k_edge_l2<32, 1><<<nb(E2), TB, 0, stream>>>(y, a1, bn1, e2w1, e2b1, nullptr, nullptr,
                                              a2, bn2, colI, hB, E2);

  // GCN3: 32 -> 16, relu  (hB -> hA)
  k_zero<<<nb((long long)N * 32), TB, 0, stream>>>(agg, N * 32);
  k_gcn_scatter<<<nb((long long)E * 32), TB, 0, stream>>>(hB, esrc, edst, degO, agg, E, 32);
  k_gcn_linear<<<nb((long long)N * 16), TB, 0, stream>>>(agg, degI, W3, b3, hA, N, 32, 16, 1);

  // GCN4: 16 -> 18, no relu  (hA -> d_out)
  k_zero<<<nb((long long)N * 16), TB, 0, stream>>>(agg, N * 16);
  k_gcn_scatter<<<nb((long long)E * 16), TB, 0, stream>>>(hA, esrc, edst, degO, agg, E, 16);
  k_gcn_linear<<<nb((long long)N * 18), TB, 0, stream>>>(agg, degI, W4, b4, (float*)d_out,
                                                         N, 16, 18, 0);
}
